// GeometryGNN_54657753809375
// MI455X (gfx1250) — matrix operand order repaired and checked by reading, not compiled, not run on hardware
//
#include <hip/hip_runtime.h>
#include <stdint.h>

// ---------------------------------------------------------------------------
// Fused GeometryGNN for MI455X (gfx1250, wave32, WMMA bf16, async-to-LDS).
//
// Math: with GIN eps=0 on K3, h = x + (sum - x) = sum(nodes) -> all three node
// features are identical from layer 1 on. The net reduces to 8 dense matmul
// stages on one H=256 vector per sample:
//   e0 = relu(x W0^T + b0)                  [K=16, padded to 32]
//   s  = 3*relu(e0 W1^T + b1) + sum(init)   [K=256]
//   per GIN layer l: u = relu(s Wa^T + ba); s = k*relu(u Wb^T + bb), k=3 (l<2) else 1
//   out[b, 0..2] = dec_w . s + dec_b
//
// Pipeline: prep kernels convert all weights f32->bf16 into d_ws once; the
// main kernel double-buffers weights in LDS and fetches the next stage with
// global_load_async_to_lds_b128 (ASYNCcnt) while WMMAs run on the current one.
// LDS: 2 x 128KB weight buffers + 2 x 32KB activation buffers = 320KB (WGP max).
// ---------------------------------------------------------------------------

typedef __attribute__((ext_vector_type(16))) __bf16 v16bf;
typedef __attribute__((ext_vector_type(8)))  float  v8f;

#define HD    256
#define ROWS  64
#define TPB   256

// LDS byte offsets (total = 327680 = 320 KB)
#define WB0_OFF   0u
#define WB1_OFF   131072u
#define ACT0_OFF  262144u
#define ACT1_OFF  294912u
#define XPAD_OFF  ACT1_OFF          // x staging aliases act1 (dead by the time act1 is written)
#define SMEM_BYTES 327680u

// d_ws element offsets (uint16 units)
#define WS_W0PAD  0u                       // [256][32] padded enc_w0
#define WS_MAT(k) (8192u + (k) * 65536u)   // 7 x [256][256] bf16 matrices

__device__ __forceinline__ uint16_t f2bf(float f) {
    uint32_t u = __float_as_uint(f);
    u += 0x7FFFu + ((u >> 16) & 1u);            // round-to-nearest-even
    return (uint16_t)(u >> 16);
}
__device__ __forceinline__ float bf2f(uint16_t h) {
    return __uint_as_float(((uint32_t)h) << 16);
}
__device__ __forceinline__ uint32_t pack2(float a, float b) {
    return (uint32_t)f2bf(a) | ((uint32_t)f2bf(b) << 16);
}

union Frag { v16bf v; uint4 q[2]; };

// A fragment (16-bit A, 16x32): lanes 0-15 row M=lane, K {k0..k0+7, k0+16..k0+23};
// lanes 16-31 row M=lane-16, K {k0+8..k0+15, k0+24..k0+31}. Two ds_load_b128.
__device__ __forceinline__ v16bf load_a(const uint16_t* base, int stride,
                                        int mbase, int k0, int lane) {
    const uint16_t* p = base + (mbase + (lane & 15)) * stride + k0 + (lane >> 4) * 8;
    Frag f;
    f.q[0] = *(const uint4*)p;
    f.q[1] = *(const uint4*)(p + 16);
    return f.v;
}

// B fragment (32x16, B[k][n] = W[n][k]): lane L holds W[nbase+(L&15)][k0+(L>>4)*8 .. +7]
// and W[nbase+(L&15)][k0+16+(L>>4)*8 .. +7], matching the A fragment K order.
__device__ __forceinline__ v16bf load_b(const uint16_t* w, int stride,
                                        int nbase, int k0, int lane) {
    const uint16_t* p = w + (nbase + (lane & 15)) * stride + k0 + (lane >> 4) * 8;
    Frag f;
    f.q[0] = *(const uint4*)p;
    f.q[1] = *(const uint4*)(p + 16);
    return f.v;
}

// Async copy of one 128KB bf16 weight matrix from d_ws into an LDS buffer.
// Each wave moves 16KB with 32 global_load_async_to_lds_b128 (512B per instr).
__device__ __forceinline__ void async_copy_w(const uint16_t* gsrc,
                                             uint32_t lds_byte_off, int tid) {
    const int      lane = tid & 31;
    const int      wave = tid >> 5;
    const uint64_t base = (uint64_t)(uintptr_t)gsrc;
    const uint32_t wo   = (uint32_t)wave * 16384u + (uint32_t)lane * 16u;
#pragma unroll 4
    for (int j = 0; j < 32; ++j) {
        uint32_t goff = wo + (uint32_t)j * 512u;
        uint32_t loff = lds_byte_off + goff;
        asm volatile("global_load_async_to_lds_b128 %0, %1, %2"
                     :: "v"(loff), "v"(goff), "s"(base) : "memory");
    }
}

__device__ __forceinline__ void wait_async_and_sync() {
    asm volatile("s_wait_asynccnt 0x0" ::: "memory");
    __syncthreads();
}

// One matmul stage over a 64x256 activation tile:
//   out = scale*relu(A * W^T + bias) (+ sum(init_nodes) if init_g).
// Each wave: 1 M-tile x 8 N-tiles; K in steps of 32 (8 WMMAs per k-step).
__device__ __forceinline__ void mm_stage(uint16_t* lds,
                                         uint32_t a_off, int astride,
                                         uint32_t w_off, int wstride, int ksteps,
                                         const float* __restrict__ bias_g,
                                         const float* __restrict__ init_g,  // nullptr or global [3*256]
                                         float scale,
                                         uint32_t out_off, int tid) {
    const int lane  = tid & 31;
    const int wave  = tid >> 5;
    const int mbase = (wave >> 1) * 16;        // 4 M-tiles, 2 waves each
    const int nhalf = (wave & 1) * 8;          // 8 N-tiles per wave

    const uint16_t* A = lds + (a_off >> 1);
    const uint16_t* W = lds + (w_off >> 1);
    uint16_t* O       = lds + (out_off >> 1);

    v8f acc[8] = {};

    for (int ks = 0; ks < ksteps; ++ks) {
        const int k0 = ks * 32;
        v16bf a = load_a(A, astride, mbase, k0, lane);
#pragma unroll
        for (int n = 0; n < 8; ++n) {
            v16bf b = load_b(W, wstride, (nhalf + n) * 16, k0, lane);
            acc[n] = __builtin_amdgcn_wmma_f32_16x16x32_bf16(
                false, a, false, b, (short)0, acc[n], false, false);
        }
    }

    // C/D layout: VGPR r -> row mbase + r + (lane>=16)*8, col nbase + (lane&15).
    const int colq = lane & 15;
    const int r0   = mbase + (lane >> 4) * 8;
#pragma unroll
    for (int n = 0; n < 8; ++n) {
        const int   c    = (nhalf + n) * 16 + colq;
        const float bias = bias_g[c];
        const float iadd = init_g ? (init_g[c] + init_g[HD + c] + init_g[2 * HD + c])
                                  : 0.0f;
#pragma unroll
        for (int r = 0; r < 8; ++r) {
            float h = acc[n][r] + bias;
            h = fmaxf(h, 0.0f);
            h = h * scale + iadd;
            O[(r0 + r) * HD + c] = f2bf(h);
        }
    }
}

// ----------------------------- prep kernels --------------------------------

// enc_w0 [256][16] f32 -> [256][32] bf16, K 16..31 zeroed.
__global__ void prep_w0(const float* __restrict__ src, uint16_t* __restrict__ dst) {
    const int t = threadIdx.x;
    const float4* s4 = (const float4*)src;
    for (int i = t; i < HD * 16 / 4; i += TPB) {
        float4 f = s4[i];
        int row = i >> 2, part = i & 3;
        *(uint2*)(dst + row * 32 + part * 4) = make_uint2(pack2(f.x, f.y), pack2(f.z, f.w));
    }
    for (int i = t; i < HD * 16 / 4; i += TPB) {
        int row = i >> 2, part = i & 3;
        *(uint2*)(dst + row * 32 + 16 + part * 4) = make_uint2(0u, 0u);
    }
}

// [256][256] f32 -> bf16 (grid = 64 blocks x 256 threads).
__global__ void prep_w(const float* __restrict__ src, uint16_t* __restrict__ dst) {
    const int i = blockIdx.x * TPB + threadIdx.x;     // 16384 float4
    float4 f = ((const float4*)src)[i];
    ((uint2*)dst)[i] = make_uint2(pack2(f.x, f.y), pack2(f.z, f.w));
}

// ------------------------------ main kernel --------------------------------

__global__ __launch_bounds__(TPB, 1)
void geo_gnn_fused(const float* __restrict__ x,
                   const uint16_t* __restrict__ wsbf,
                   const float* __restrict__ eb0, const float* __restrict__ eb1,
                   const float* __restrict__ init_nodes,
                   const float* __restrict__ gb1, const float* __restrict__ gb2,
                   const float* __restrict__ dw,  const float* __restrict__ db,
                   float* __restrict__ out) {
    extern __shared__ char smem[];
    uint16_t* lds = (uint16_t*)smem;

    const int tid   = threadIdx.x;
    const int rbase = blockIdx.x * ROWS;

    // --- stage -1 staging: x tile -> XPAD bf16 (zero-pad K 16..31), w0pad -> WB0
    {
        uint16_t* xpad = lds + (XPAD_OFF >> 1);
        const float4* xsrc = (const float4*)(x + (size_t)rbase * 16);
        for (int i = tid; i < ROWS * 16 / 4; i += TPB) {   // 256 float4
            float4 f = xsrc[i];
            int row = i >> 2, part = i & 3;
            *(uint2*)(xpad + row * 32 + part * 4) = make_uint2(pack2(f.x, f.y), pack2(f.z, f.w));
        }
        for (int i = tid; i < ROWS * 16 / 4; i += TPB) {
            int row = i >> 2, part = i & 3;
            *(uint2*)(xpad + row * 32 + 16 + part * 4) = make_uint2(0u, 0u);
        }
        // w0pad: 16KB bf16 sync copy (already converted in ws)
        uint16_t* wb0 = lds + (WB0_OFF >> 1);
        const uint4* src = (const uint4*)(wsbf + WS_W0PAD);
        for (int i = tid; i < (HD * 32) / 8; i += TPB)     // 1024 x 16B
            ((uint4*)wb0)[i] = src[i];
    }

    // Kick async fetch of stage-0 weights (ew1) into WB1 while enc0 runs.
    async_copy_w(wsbf + WS_MAT(0), WB1_OFF, tid);
    __syncthreads();

    // enc0: relu(x W0^T + b0) -> ACT0   [K padded to 32]
    mm_stage(lds, XPAD_OFF, 32, WB0_OFF, 32, 1, eb0, nullptr, 1.0f, ACT0_OFF, tid);
    wait_async_and_sync();

    // 7 dense H x H stages with double-buffered weights.
    const float* biasv[7] = { eb1, gb1, gb2, gb1 + HD, gb2 + HD, gb1 + 2 * HD, gb2 + 2 * HD };
    const float  sclv[7]  = { 3.0f, 1.0f, 3.0f, 1.0f, 3.0f, 1.0f, 1.0f };

    for (int s = 0; s < 7; ++s) {
        const uint32_t wbuf  = (s & 1) ? WB0_OFF : WB1_OFF;    // stage s weights live here
        const uint32_t wnext = (s & 1) ? WB1_OFF : WB0_OFF;
        if (s < 6) async_copy_w(wsbf + WS_MAT(s + 1), wnext, tid);

        const uint32_t a_off = (s & 1) ? ACT1_OFF : ACT0_OFF;
        const uint32_t o_off = (s & 1) ? ACT0_OFF : ACT1_OFF;
        const float*   ig    = (s == 0) ? init_nodes : nullptr;

        mm_stage(lds, a_off, HD, wbuf, HD, 8, biasv[s], ig, sclv[s], o_off, tid);
        wait_async_and_sync();
    }

    // Decoder: one dot product per row (ACT1 holds final features);
    // all 3 node outputs are identical.
    const float bd = db[0];
    const uint16_t* act1 = lds + (ACT1_OFF >> 1);
    for (int r = tid; r < ROWS; r += TPB) {
        const uint16_t* vr = act1 + r * HD;
        float s = bd;
#pragma unroll 8
        for (int c = 0; c < HD; ++c) s += dw[c] * bf2f(vr[c]);
        const size_t row = (size_t)(rbase + r) * 3;
        out[row + 0] = s;
        out[row + 1] = s;
        out[row + 2] = s;
    }
}

extern "C" void kernel_launch(void* const* d_in, const int* in_sizes, int n_in,
                              void* d_out, int out_size, void* d_ws, size_t ws_size,
                              hipStream_t stream) {
    const float* x    = (const float*)d_in[0];
    const float* ew0  = (const float*)d_in[1];
    const float* eb0  = (const float*)d_in[2];
    const float* ew1  = (const float*)d_in[3];
    const float* eb1  = (const float*)d_in[4];
    const float* ini  = (const float*)d_in[5];
    const float* gw1  = (const float*)d_in[6];
    const float* gb1  = (const float*)d_in[7];
    const float* gw2  = (const float*)d_in[8];
    const float* gb2  = (const float*)d_in[9];
    const float* dw   = (const float*)d_in[10];
    const float* db   = (const float*)d_in[11];
    float* out        = (float*)d_out;
    uint16_t* ws16    = (uint16_t*)d_ws;

    // Prep: convert all weights to bf16 once into d_ws (~912 KB, L2-resident).
    prep_w0<<<dim3(1), dim3(TPB), 0, stream>>>(ew0, ws16 + WS_W0PAD);
    const float* mats[7] = { ew1,
                             gw1,              gw2,
                             gw1 + 1 * HD * HD, gw2 + 1 * HD * HD,
                             gw1 + 2 * HD * HD, gw2 + 2 * HD * HD };
    for (int k = 0; k < 7; ++k)
        prep_w<<<dim3(64), dim3(TPB), 0, stream>>>(mats[k], ws16 + WS_MAT(k));

    const int B       = in_sizes[0] / 16;   // 32768
    const int nblocks = B / ROWS;           // 512
    geo_gnn_fused<<<dim3(nblocks), dim3(TPB), SMEM_BYTES, stream>>>(
        x, ws16, eb0, eb1, ini, gb1, gb2, dw, db, out);
}
